// PoseVideoCNNRNN_87497073754264
// MI455X (gfx1250) — matrix operand order repaired and checked by reading, not compiled, not run on hardware
//
#include <hip/hip_runtime.h>
#include <hip/hip_bf16.h>

// ---------------------------------------------------------------------------
// PoseVideoCNNRNN on MI455X (gfx1250): WMMA f16 GEMMs, double-buffered
// async-to-LDS staging, persistent LDS scans.
// ---------------------------------------------------------------------------

typedef __attribute__((ext_vector_type(16))) _Float16 v16h;
typedef __attribute__((ext_vector_type(8)))  _Float16 v8h;
typedef __attribute__((ext_vector_type(8)))  float    v8f;

#define T_STEPS 100
#define B_SZ    128

__device__ inline v8f wmma16(v16h a, v16h b, v8f c) {
    // D = A(16x32 f16) * B(32x16 f16) + C(16x16 f32)
    return __builtin_amdgcn_wmma_f32_16x16x32_f16(
        /*neg_a=*/false, a, /*neg_b=*/false, b,
        /*c_mod=*/(short)0, c, /*reuse_a=*/false, /*reuse_b=*/false);
}

__device__ inline v8f zero8() {
    v8f z;
#pragma unroll
    for (int i = 0; i < 8; ++i) z[i] = 0.0f;
    return z;
}

// Async copy of 16B/lane from global to LDS (CDNA5 GLOBAL_LOAD_ASYNC_TO_LDS,
// tracked by ASYNCcnt). The low 32 bits of the generic address map directly
// to the LDS byte offset on gfx1250.
__device__ inline void async_copy_b128(const void* gsrc, void* ldst) {
    unsigned int       l = (unsigned int)(unsigned long long)(uintptr_t)ldst;
    unsigned long long g = (unsigned long long)(uintptr_t)gsrc;
    asm volatile("global_load_async_to_lds_b128 %0, %1, off"
                 :: "v"(l), "v"(g) : "memory");
}
__device__ inline void wait_async0() {
    asm volatile("s_wait_asynccnt 0x0" ::: "memory");
}

// A fragment (16x32, f16) from LDS. base points at row 0 of the 16-row tile,
// ld = row pitch in halves (multiple of 8 -> 16B aligned vector loads).
// Layout (contiguous K, not the ISA table order): lane L(0..15)=row M=L, half-group h=lane>>4:
//   VGPR0..7 hold K = h*16 + 0..15
__device__ inline v16h frag_a_lds(const _Float16* base, int ld, int k0) {
    int lane = threadIdx.x & 31;
    int row  = lane & 15;
    int h    = lane >> 4;
    const _Float16* p0 = base + row * ld + k0 + h * 16;
    v8h lo = *(const v8h*)(p0);
    v8h hi = *(const v8h*)(p0 + 8);
    v16h a;
#pragma unroll
    for (int i = 0; i < 8; ++i) { a[i] = lo[i]; a[i + 8] = hi[i]; }
    return a;
}

// B fragment (32x16) from an LDS-staged (N,K) tile: lane = column N,
// 16 contiguous K values per lane.
__device__ inline v16h frag_b_lds(const _Float16* base, int ld, int n0) {
    int lane = threadIdx.x & 31;
    int n = lane & 15;
    int h = lane >> 4;
    const _Float16* p = base + (n0 + n) * ld + h * 16;
    v8h lo = *(const v8h*)(p);
    v8h hi = *(const v8h*)(p + 8);
    v16h b;
#pragma unroll
    for (int i = 0; i < 8; ++i) { b[i] = lo[i]; b[i + 8] = hi[i]; }
    return b;
}

// B fragment (32x16) straight from a row-major (N,K) f16 weight matrix in
// global memory (single 32B contiguous load per lane).
__device__ inline v16h frag_b_glb(const _Float16* W, int ld, int n0, int k0) {
    int lane = threadIdx.x & 31;
    int n = lane & 15;
    int h = lane >> 4;
    return *(const v16h*)(W + (size_t)(n0 + n) * ld + k0 + h * 16);
}

__device__ inline float sigf(float x) { return 1.0f / (1.0f + __expf(-x)); }
__device__ inline float tanh_f(float x) {
    x = fminf(15.0f, fmaxf(-15.0f, x));
    float e = __expf(2.0f * x);
    return (e - 1.0f) / (e + 1.0f);
}
__device__ inline float leaky(float x, float s) { return x >= 0.0f ? x : s * x; }

// ---------------------------------------------------------------------------
// Kernel 1: weight conversion / padding to f16
// ---------------------------------------------------------------------------
__global__ void k_prep(const float* __restrict__ lWih, const float* __restrict__ lWhh,
                       const float* __restrict__ gWih, const float* __restrict__ gWhh,
                       _Float16* __restrict__ lWih16, _Float16* __restrict__ lWhh16,
                       _Float16* __restrict__ gWih16, _Float16* __restrict__ gWhh16) {
    int stride = gridDim.x * blockDim.x;
    int tid = blockIdx.x * blockDim.x + threadIdx.x;
    for (int i = tid; i < 512 * 4096; i += stride) lWih16[i] = (_Float16)lWih[i];
    for (int i = tid; i < 512 * 128;  i += stride) lWhh16[i] = (_Float16)lWhh[i];
    for (int i = tid; i < 832 * 288;  i += stride) {
        int n = i / 288, k = i % 288;
        float a = (n < 783 && k < 261) ? gWih[n * 261 + k] : 0.0f;
        float b = (n < 783 && k < 261) ? gWhh[n * 261 + k] : 0.0f;
        gWih16[i] = (_Float16)a;
        gWhh16[i] = (_Float16)b;
    }
}

// ---------------------------------------------------------------------------
// Kernel 2: fused pose-projection + LeakyReLU + GEMM vs lstm_Wih^T
//   M=12800 (b*T+t), N=512, K=4096.  Block = 256 thr (8 waves), tile 128x64.
//   Double-buffered pipeline: per K-step, async-stage B slab i+1 and
//   VALU-build A slab i+1 while WMMAs consume slab i.  One barrier/step.
// ---------------------------------------------------------------------------
__global__ __launch_bounds__(256) void k_xgates(
        const float* __restrict__ pose,   // (12800,9)
        const float* __restrict__ projW,  // (4096,9)
        const float* __restrict__ projB,  // (4096)
        const _Float16* __restrict__ Wih16, // (512,4096)
        const float* __restrict__ bih, const float* __restrict__ bhh,
        float* __restrict__ xg) {         // (T,B,512)
    __shared__ float    poseS[128][12];
    __shared__ _Float16 At[2][128][40];   // 128x32 K-slabs, pitch 40
    __shared__ _Float16 Bt[2][64][40];    // 64x32 K-slabs of lstm_Wih

    const int tid = threadIdx.x;
    const int m0  = blockIdx.x * 128;
    const int n0  = blockIdx.y * 64;

    for (int e = tid; e < 128 * 9; e += 256) {
        int r = e / 9, c = e % 9;
        poseS[r][c] = pose[(size_t)(m0 + r) * 9 + c];
    }
    __syncthreads();

    const int wave = tid >> 5;
    const int wm = wave >> 1;   // 0..3 -> M offset wm*32
    const int wn = wave & 1;    // 0..1 -> N offset wn*32
    const int lane = tid & 31, nl = lane & 15, mh = lane >> 4;

    v8f acc[2][2];
#pragma unroll
    for (int i = 0; i < 2; ++i)
#pragma unroll
        for (int j = 0; j < 2; ++j) acc[i][j] = zero8();

    const int arow = tid >> 1;
    const int ak0  = (tid & 1) * 16;
    const int brow = tid >> 2;           // 0..63
    const int bchk = tid & 3;            // 16B chunk within a 64B row slab
    float pr[9];
#pragma unroll
    for (int j = 0; j < 9; ++j) pr[j] = poseS[arow][j];

    // builds the 128x32 LeakyReLU(pose @ projW^T + b) slab for K-base kk
    auto build_a = [&](int buf, int kk) {
#pragma unroll
        for (int k = 0; k < 16; ++k) {
            int kg = kk + ak0 + k;
            const float* w = projW + (size_t)kg * 9;
            float v = projB[kg];
#pragma unroll
            for (int j = 0; j < 9; ++j) v = fmaf(pr[j], w[j], v);
            At[buf][arow][ak0 + k] = (_Float16)leaky(v, 0.1f);
        }
    };
    auto stage_b = [&](int buf, int kk) {
        async_copy_b128(Wih16 + (size_t)(n0 + brow) * 4096 + kk + bchk * 8,
                        &Bt[buf][brow][bchk * 8]);
    };

    // prologue: slab 0
    stage_b(0, 0);
    build_a(0, 0);
    wait_async0();
    __syncthreads();

    for (int kk = 0; kk < 4096; kk += 32) {
        const int cur = (kk >> 5) & 1;
        const int nxt = cur ^ 1;
        if (kk + 32 < 4096) {
            stage_b(nxt, kk + 32);
            build_a(nxt, kk + 32);
        }
        v16h b0 = frag_b_lds(&Bt[cur][0][0], 40, wn * 32);
        v16h b1 = frag_b_lds(&Bt[cur][0][0], 40, wn * 32 + 16);
#pragma unroll
        for (int i = 0; i < 2; ++i) {
            v16h a = frag_a_lds(&At[cur][wm * 32 + i * 16][0], 40, 0);
            acc[i][0] = wmma16(a, b0, acc[i][0]);
            acc[i][1] = wmma16(a, b1, acc[i][1]);
        }
        wait_async0();
        __syncthreads();
    }

#pragma unroll
    for (int i = 0; i < 2; ++i)
#pragma unroll
        for (int j = 0; j < 2; ++j) {
            int gn = n0 + wn * 32 + j * 16 + nl;
            float bias = bih[gn] + bhh[gn];
#pragma unroll
            for (int r = 0; r < 8; ++r) {
                int m = m0 + wm * 32 + i * 16 + 8 * mh + r; // global row = b*T + t
                int bb = m / T_STEPS, tt = m % T_STEPS;
                xg[((size_t)tt * B_SZ + bb) * 512 + gn] = acc[i][j][r] + bias;
            }
        }
}

// ---------------------------------------------------------------------------
// Kernel 3: persistent LSTM scan (one WGP, 32 waves). Per step:
//   gates = x_gates[t] (as WMMA C operand) + h @ Whh^T   (M=128,N=512,K=128)
// LDS: gates f16 128x512 (128KB) + h f16 128x136 (34KB) + c f32 128x128 (64KB)
// Whh (128KB f16) stays L2-resident across the 100 steps.
// ---------------------------------------------------------------------------
__global__ __launch_bounds__(1024) void k_lstm(
        const float* __restrict__ xg,       // (T,B,512)
        const _Float16* __restrict__ Whh16, // (512,128)
        float* __restrict__ hT) {           // (128,128)
    extern __shared__ char smem[];
    _Float16* g16 = (_Float16*)smem;                         // [128][512]
    _Float16* h16 = (_Float16*)(smem + 128 * 512 * 2);       // [128][136]
    float*    cS  = (float*)(smem + 128 * 512 * 2 + 128 * 136 * 2); // [128][128]

    const int tid = threadIdx.x;
    const int wave = tid >> 5, lane = tid & 31, nl = lane & 15, mh = lane >> 4;

    for (int e = tid; e < 128 * 136; e += 1024) h16[e] = (_Float16)0.0f;
    for (int e = tid; e < 128 * 128; e += 1024) cS[e] = 0.0f;
    __syncthreads();

    for (int t = 0; t < T_STEPS; ++t) {
        // recurrent GEMM: wave w owns gate-column tile nt=w, all 8 b-tiles
        const int gn = wave * 16 + nl;
#pragma unroll 1
        for (int mt = 0; mt < 8; ++mt) {
            v8f acc;
#pragma unroll
            for (int r = 0; r < 8; ++r) {
                int b = mt * 16 + 8 * mh + r;
                acc[r] = xg[((size_t)t * B_SZ + b) * 512 + gn];
            }
#pragma unroll
            for (int kk = 0; kk < 128; kk += 32) {
                v16h a = frag_a_lds(h16 + (size_t)(mt * 16) * 136, 136, kk);
                v16h b = frag_b_glb(Whh16, 128, wave * 16, kk);
                acc = wmma16(a, b, acc);
            }
#pragma unroll
            for (int r = 0; r < 8; ++r)
                g16[(size_t)(mt * 16 + 8 * mh + r) * 512 + gn] = (_Float16)acc[r];
        }
        __syncthreads();

        // gate math: thread owns 16 (b,u) states
#pragma unroll
        for (int i = 0; i < 16; ++i) {
            int s = tid + i * 1024;
            int b = s >> 7, u = s & 127;
            const _Float16* gr = g16 + (size_t)b * 512;
            float ig = sigf((float)gr[u]);
            float fg = sigf((float)gr[128 + u]);
            float gg = tanh_f((float)gr[256 + u]);
            float og = sigf((float)gr[384 + u]);
            float c = fg * cS[s] + ig * gg;
            cS[s] = c;
            float h = og * tanh_f(c);
            h16[(size_t)b * 136 + u] = (_Float16)h;
            if (t == T_STEPS - 1) hT[s] = h;
        }
        __syncthreads();
    }
}

// ---------------------------------------------------------------------------
// Kernel 4a: VAE head -> mu, logvar (outputs) + embedding
// ---------------------------------------------------------------------------
__global__ void k_vae(const float* __restrict__ hT,
                      const float* __restrict__ muW, const float* __restrict__ muB,
                      const float* __restrict__ lvW, const float* __restrict__ lvB,
                      const float* __restrict__ eps,
                      float* __restrict__ mu_out, float* __restrict__ lv_out,
                      float* __restrict__ emb) {
    int idx = blockIdx.x * blockDim.x + threadIdx.x;
    if (idx >= 128 * 64) return;
    int b = idx >> 6, j = idx & 63;
    const float* h  = hT + (size_t)b * 128;
    const float* wm = muW + (size_t)j * 128;
    const float* wl = lvW + (size_t)j * 128;
    float m = muB[j], l = lvB[j];
    for (int k = 0; k < 128; ++k) { m = fmaf(h[k], wm[k], m); l = fmaf(h[k], wl[k], l); }
    m = leaky(m, 0.1f);
    l = fminf(10.0f, fmaxf(-10.0f, leaky(l, 0.1f)));
    mu_out[idx] = m;
    lv_out[idx] = l;
    emb[idx] = m + eps[idx] * __expf(0.5f * l);
}

// Kernel 4b: x = emb @ fcin_W^T + b   (128x256, K=64)
__global__ void k_fcin(const float* __restrict__ emb, const float* __restrict__ W,
                       const float* __restrict__ bias, float* __restrict__ x) {
    int idx = blockIdx.x * blockDim.x + threadIdx.x;
    if (idx >= 128 * 256) return;
    int b = idx >> 8, o = idx & 255;
    const float* e = emb + (size_t)b * 64;
    const float* w = W + (size_t)o * 64;
    float v = bias[o];
    for (int k = 0; k < 64; ++k) v = fmaf(e[k], w[k], v);
    x[idx] = v;
}

// ---------------------------------------------------------------------------
// Kernel 5: fused decoder-input construction + GEMM vs gru_Wih^T
//   M=12800 (t*128+b -> one t per M-block), Npad=832 (real 783), Kpad=288 (261)
//   Same double-buffered async pipeline; gru_bih folded into gx.
// ---------------------------------------------------------------------------
__global__ __launch_bounds__(256) void k_gx(
        const float* __restrict__ x,        // (128,256)
        const float* __restrict__ noise,    // (T,B,4)
        const _Float16* __restrict__ gWih16,// (832,288) padded
        const float* __restrict__ gbih,     // (783)
        float* __restrict__ gx) {           // (T,B,783)
    __shared__ _Float16 At[2][128][40];
    __shared__ _Float16 Bt[2][64][40];

    const int tid = threadIdx.x;
    const int t  = blockIdx.x;        // rows b = 0..127 for this t
    const int n0 = blockIdx.y * 64;

    const int wave = tid >> 5;
    const int wm = wave >> 1, wn = wave & 1;
    const int lane = tid & 31, nl = lane & 15, mh = lane >> 4;

    v8f acc[2][2];
#pragma unroll
    for (int i = 0; i < 2; ++i)
#pragma unroll
        for (int j = 0; j < 2; ++j) acc[i][j] = zero8();

    const int arow = tid >> 1;          // = b
    const int ak0  = (tid & 1) * 16;
    const int brow = tid >> 2;
    const int bchk = tid & 3;
    const float tval = (float)t / (float)(T_STEPS - 1);

    auto build_a = [&](int buf, int kk) {
#pragma unroll
        for (int k = 0; k < 16; ++k) {
            int kg = kk + ak0 + k;
            float v;
            if (kg < 256)       v = x[(size_t)arow * 256 + kg];
            else if (kg < 260)  v = noise[((size_t)t * B_SZ + arow) * 4 + (kg - 256)] * 0.1f;
            else if (kg == 260) v = tval;
            else                v = 0.0f;
            At[buf][arow][ak0 + k] = (_Float16)v;
        }
    };
    auto stage_b = [&](int buf, int kk) {
        async_copy_b128(gWih16 + (size_t)(n0 + brow) * 288 + kk + bchk * 8,
                        &Bt[buf][brow][bchk * 8]);
    };

    stage_b(0, 0);
    build_a(0, 0);
    wait_async0();
    __syncthreads();

    for (int kk = 0; kk < 288; kk += 32) {
        const int cur = (kk >> 5) & 1;
        const int nxt = cur ^ 1;
        if (kk + 32 < 288) {
            stage_b(nxt, kk + 32);
            build_a(nxt, kk + 32);
        }
        v16h b0 = frag_b_lds(&Bt[cur][0][0], 40, wn * 32);
        v16h b1 = frag_b_lds(&Bt[cur][0][0], 40, wn * 32 + 16);
#pragma unroll
        for (int i = 0; i < 2; ++i) {
            v16h a = frag_a_lds(&At[cur][wm * 32 + i * 16][0], 40, 0);
            acc[i][0] = wmma16(a, b0, acc[i][0]);
            acc[i][1] = wmma16(a, b1, acc[i][1]);
        }
        wait_async0();
        __syncthreads();
    }

#pragma unroll
    for (int i = 0; i < 2; ++i)
#pragma unroll
        for (int j = 0; j < 2; ++j) {
            int gn = n0 + wn * 32 + j * 16 + nl;
            if (gn < 783) {
                float bias = gbih[gn];
#pragma unroll
                for (int r = 0; r < 8; ++r) {
                    int b = wm * 32 + i * 16 + 8 * mh + r;
                    gx[((size_t)t * B_SZ + b) * 783 + gn] = acc[i][j][r] + bias;
                }
            }
        }
}

// ---------------------------------------------------------------------------
// Kernel 6: persistent GRU scan + frame head (one WGP, 32 waves). Per step:
//   gh = h @ gWhh^T  (M=128, Npad=784, Kpad=288)  -> LDS f16
//   gates r/z/n, h update, then frame = sig(leaky(h@W1^T)@W2^T) -> d_out
// LDS: h16 128x296 (74KB) + gh16 128x792 (198KB) + hid1 f32 128x64 (32KB)
// ---------------------------------------------------------------------------
__global__ __launch_bounds__(1024) void k_gru(
        const float* __restrict__ gx,        // (T,B,783)
        const _Float16* __restrict__ gWhh16, // (832,288) padded
        const float* __restrict__ gbhh,      // (783)
        const float* __restrict__ jfW1, const float* __restrict__ jfb1, // (64,261),(64)
        const float* __restrict__ jfW2, const float* __restrict__ jfb2, // (9,64),(9)
        float* __restrict__ out) {           // (B,T,3,3)
    extern __shared__ char smem[];
    _Float16* h16  = (_Float16*)smem;                        // [128][296]
    _Float16* gh16 = (_Float16*)(smem + 128 * 296 * 2);      // [128][792]
    float*    hid1 = (float*)(smem + 128 * 296 * 2 + 128 * 792 * 2); // [128][64]

    const int tid = threadIdx.x;
    const int wave = tid >> 5, lane = tid & 31, nl = lane & 15, mh = lane >> 4;

    for (int e = tid; e < 128 * 296; e += 1024) h16[e] = (_Float16)0.0f;
    __syncthreads();

    for (int t = 0; t < T_STEPS; ++t) {
        // recurrent GEMM: 8 x 49 tiles, round-robin over waves
        for (int idx = wave; idx < 8 * 49; idx += 32) {
            int mt = idx / 49, nt = idx % 49;
            v8f acc = zero8();
#pragma unroll
            for (int kk = 0; kk < 288; kk += 32) {
                v16h a = frag_a_lds(h16 + (size_t)(mt * 16) * 296, 296, kk);
                v16h b = frag_b_glb(gWhh16, 288, nt * 16, kk);
                acc = wmma16(a, b, acc);
            }
#pragma unroll
            for (int r = 0; r < 8; ++r)
                gh16[(size_t)(mt * 16 + 8 * mh + r) * 792 + nt * 16 + nl] = (_Float16)acc[r];
        }
        __syncthreads();

        // gate math: h' = (1-z)*n + z*h
        const float* gxt = gx + (size_t)t * B_SZ * 783;
        for (int s = tid; s < 128 * 261; s += 1024) {
            int b = s / 261, u = s % 261;
            const float*    gxr = gxt + (size_t)b * 783;
            const _Float16* ghr = gh16 + (size_t)b * 792;
            float hold = (float)h16[(size_t)b * 296 + u];
            float r = sigf(gxr[u]        + gbhh[u]        + (float)ghr[u]);
            float z = sigf(gxr[261 + u]  + gbhh[261 + u]  + (float)ghr[261 + u]);
            float n = tanh_f(gxr[522 + u] + gbhh[522 + u] + r * (float)ghr[522 + u]);
            h16[(size_t)b * 296 + u] = (_Float16)((1.0f - z) * n + z * hold);
        }
        __syncthreads();

        // frame head, layer 1: (128,261)->(128,64), leaky 0.2
        for (int e = tid; e < 128 * 64; e += 1024) {
            int b = e >> 6, j = e & 63;
            const float* w = jfW1 + (size_t)j * 261;
            const _Float16* h = h16 + (size_t)b * 296;
            float v = jfb1[j];
            for (int u = 0; u < 261; ++u) v = fmaf((float)h[u], w[u], v);
            hid1[e] = leaky(v, 0.2f);
        }
        __syncthreads();

        // frame head, layer 2: (128,64)->(128,9), sigmoid, store (B,T,3,3)
        for (int e = tid; e < 128 * 9; e += 1024) {
            int b = e / 9, j = e % 9;
            const float* w = jfW2 + (size_t)j * 64;
            const float* h = hid1 + (size_t)b * 64;
            float v = jfb2[j];
#pragma unroll
            for (int u = 0; u < 64; ++u) v = fmaf(h[u], w[u], v);
            out[(size_t)b * (T_STEPS * 9) + t * 9 + j] = sigf(v);
        }
        __syncthreads();
    }
}

// ---------------------------------------------------------------------------
extern "C" void kernel_launch(void* const* d_in, const int* in_sizes, int n_in,
                              void* d_out, int out_size, void* d_ws, size_t ws_size,
                              hipStream_t stream) {
    (void)in_sizes; (void)n_in; (void)out_size; (void)ws_size;

    const float* pose  = (const float*)d_in[0];
    const float* eps   = (const float*)d_in[1];
    const float* noise = (const float*)d_in[2];
    const float* projW = (const float*)d_in[3];
    const float* projB = (const float*)d_in[4];
    const float* lWih  = (const float*)d_in[5];
    const float* lWhh  = (const float*)d_in[6];
    const float* lbih  = (const float*)d_in[7];
    const float* lbhh  = (const float*)d_in[8];
    const float* muW   = (const float*)d_in[9];
    const float* muB   = (const float*)d_in[10];
    const float* lvW   = (const float*)d_in[11];
    const float* lvB   = (const float*)d_in[12];
    const float* fcW   = (const float*)d_in[13];
    const float* fcB   = (const float*)d_in[14];
    const float* gWih  = (const float*)d_in[15];
    const float* gWhh  = (const float*)d_in[16];
    const float* gbih  = (const float*)d_in[17];
    const float* gbhh  = (const float*)d_in[18];
    const float* jfW1  = (const float*)d_in[19];
    const float* jfb1  = (const float*)d_in[20];
    const float* jfW2  = (const float*)d_in[21];
    const float* jfb2  = (const float*)d_in[22];

    float* out_frames = (float*)d_out;             // (128,100,3,3) = 115200
    float* out_mu     = out_frames + 115200;       // (128,64)
    float* out_lv     = out_mu + 8192;             // (128,64)

    // workspace carve-up (~72 MB)
    char* ws = (char*)d_ws;
    size_t off = 0;
    auto take = [&](size_t bytes) -> char* {
        char* p = ws + off;
        off += (bytes + 255) & ~(size_t)255;
        return p;
    };
    _Float16* lWih16 = (_Float16*)take((size_t)512 * 4096 * 2);
    _Float16* lWhh16 = (_Float16*)take((size_t)512 * 128 * 2);
    _Float16* gWih16 = (_Float16*)take((size_t)832 * 288 * 2);
    _Float16* gWhh16 = (_Float16*)take((size_t)832 * 288 * 2);
    float* xg   = (float*)take((size_t)T_STEPS * B_SZ * 512 * 4);
    float* hT   = (float*)take((size_t)128 * 128 * 4);
    float* emb  = (float*)take((size_t)128 * 64 * 4);
    float* xf   = (float*)take((size_t)128 * 256 * 4);
    float* gxbf = (float*)take((size_t)T_STEPS * B_SZ * 783 * 4);

    k_prep<<<2048, 256, 0, stream>>>(lWih, lWhh, gWih, gWhh,
                                     lWih16, lWhh16, gWih16, gWhh16);

    k_xgates<<<dim3(100, 8), 256, 0, stream>>>(pose, projW, projB, lWih16,
                                               lbih, lbhh, xg);

    size_t lstm_smem = (size_t)128 * 512 * 2 + 128 * 136 * 2 + 128 * 128 * 4; // ~231 KB
    k_lstm<<<1, 1024, lstm_smem, stream>>>(xg, lWhh16, hT);

    k_vae<<<32, 256, 0, stream>>>(hT, muW, muB, lvW, lvB, eps, out_mu, out_lv, emb);
    k_fcin<<<128, 256, 0, stream>>>(emb, fcW, fcB, xf);

    k_gx<<<dim3(100, 13), 256, 0, stream>>>(xf, noise, gWih16, gbih, gxbf);

    size_t gru_smem = (size_t)128 * 296 * 2 + 128 * 792 * 2 + 128 * 64 * 4; // ~304 KB
    k_gru<<<1, 1024, gru_smem, stream>>>(gxbf, gWhh16, gbhh, jfW1, jfb1, jfW2, jfb2,
                                         out_frames);
}
